// Mamba3Block_15290083574004
// MI455X (gfx1250) — compile-verified, matrix operand order checked by reading, not run on hardware
//
#include <hip/hip_runtime.h>
#include <hip/hip_bf16.h>
#include <math.h>

typedef __attribute__((ext_vector_type(16))) __bf16 v16bf;
typedef __attribute__((ext_vector_type(8)))  __bf16 v8bf;
typedef __attribute__((ext_vector_type(8)))  float  v8f;
typedef __attribute__((ext_vector_type(4)))  unsigned int v4u;
typedef __attribute__((ext_vector_type(8)))  int v8i;
typedef __attribute__((ext_vector_type(4)))  int v4i;

typedef unsigned short bf16r;   // raw bf16 storage

static constexpr int B_ = 2, S_ = 1024, D_ = 768;
static constexpr int HEADDIM = 64, DSTATE = 64;
static constexpr int DINNER = 1536, NHEADS = 24, DMLP = 1920;
static constexpr int PROJ = 2 * DINNER + 2 * NHEADS * DSTATE + NHEADS + NHEADS * (DSTATE / 2); // 6936
static constexpr int TOK = B_ * S_;           // 2048
static constexpr int PROJ_PAD = 7040;         // 55 * 128
static constexpr int OFF_Z  = 0;
static constexpr int OFF_X  = DINNER;                       // 1536
static constexpr int OFF_B  = 2 * DINNER;                   // 3072
static constexpr int OFF_C  = 2 * DINNER + NHEADS * DSTATE; // 4608
static constexpr int OFF_DT = 2 * DINNER + 2 * NHEADS * DSTATE; // 6144
static constexpr int OFF_TH = OFF_DT + NHEADS;              // 6168

#if __has_builtin(__builtin_amdgcn_tensor_load_to_lds)
#define USE_TDM 1
#else
#define USE_TDM 0
#endif

__device__ __forceinline__ bf16r f2bf(float f) {
    unsigned int u = __float_as_uint(f);
    unsigned int r = u + 0x7FFFu + ((u >> 16) & 1u);   // round-to-nearest-even
    return (bf16r)(r >> 16);
}

// ---------------------------------------------------------------------------
// fp32 -> bf16 weight convert with row padding (pad rows are zero)
// ---------------------------------------------------------------------------
__global__ void k_cvt_pad(const float* __restrict__ w, bf16r* __restrict__ o,
                          int rows, int cols, int rowsPad) {
    size_t i = (size_t)blockIdx.x * 256 + threadIdx.x;
    size_t tot = (size_t)rowsPad * cols;
    if (i >= tot) return;
    int r = (int)(i / cols);
    o[i] = (r < rows) ? f2bf(w[i]) : (bf16r)0;
}

// ---------------------------------------------------------------------------
// RMSNorm (fp32 in) -> bf16 out, one block per row
// ---------------------------------------------------------------------------
__global__ __launch_bounds__(256) void k_rmsnorm_bf16(
    const float* __restrict__ x, const float* __restrict__ w,
    bf16r* __restrict__ o, int D) {
    int row = blockIdx.x;
    const float* xr = x + (size_t)row * D;
    float ss = 0.f;
    for (int i = threadIdx.x; i < D; i += 256) { float v = xr[i]; ss += v * v; }
    __shared__ float red[256];
    red[threadIdx.x] = ss;
    __syncthreads();
    for (int s = 128; s > 0; s >>= 1) {
        if (threadIdx.x < s) red[threadIdx.x] += red[threadIdx.x + s];
        __syncthreads();
    }
    float scale = rsqrtf(red[0] / (float)D + 1e-5f);
    bf16r* orow = o + (size_t)row * D;
    for (int i = threadIdx.x; i < D; i += 256)
        orow[i] = f2bf(xr[i] * scale * w[i]);
}

// ---------------------------------------------------------------------------
// Tensor Data Mover: DMA a 2D tile (tileW x tileH bf16, row stride strideElems)
// from global memory into LDS at byte offset lds_off. Issued once per calling
// wave; completion tracked by TENSORcnt. D# layout per CDNA5 ISA ch.8.
// 6-arg builtin form (clang-23 / therock): (v4u, v8i, v4i, v4i, v8i, i32 cpol)
// ---------------------------------------------------------------------------
#if USE_TDM
__device__ __forceinline__ void tdm_load_2d_bf16(
    const bf16r* gptr, unsigned lds_off,
    unsigned tileW, unsigned tileH, unsigned strideElems) {
    unsigned long long ga = (unsigned long long)gptr;
    v4u g0;
    g0[0] = 1u;                                   // count=1, user descriptor
    g0[1] = lds_off;                              // lds_addr (bytes)
    g0[2] = (unsigned)(ga & 0xFFFFFFFFu);         // global_addr[31:0]
    g0[3] = (unsigned)((ga >> 32) & 0x01FFFFFFu)  // global_addr[56:32]
            | (2u << 30);                         // type = 2 (image)
    const unsigned td0 = 1u << 20;                // tensor_dim0 (all in-bounds)
    const unsigned td1 = 1u << 20;                // tensor_dim1
    v8i g1;
    g1[0] = (int)(1u << 16);                      // data_size=1 (2 bytes), mask=0
    g1[1] = (int)((td0 & 0xFFFFu) << 16);         // [47:32]=atomic_addr=0, [63:48]=td0 lo
    g1[2] = (int)((td0 >> 16) | ((td1 & 0xFFFFu) << 16));   // td0 hi | td1 lo
    g1[3] = (int)((td1 >> 16) | (tileW << 16));   // td1 hi | tile_dim0
    g1[4] = (int)(tileH & 0xFFFFu);               // tile_dim1 (tile_dim2 = 0)
    g1[5] = (int)strideElems;                     // tensor_dim0_stride[31:0]
    g1[6] = 0;                                    // stride hi, dim1_stride lo
    g1[7] = 0;
    v4i g2 = {0, 0, 0, 0};
    v4i g3 = {0, 0, 0, 0};
    v8i g4 = {0, 0, 0, 0, 0, 0, 0, 0};
    __builtin_amdgcn_tensor_load_to_lds(g0, g1, g2, g3, g4, 0);
}
#endif

// ---------------------------------------------------------------------------
// bf16 WMMA GEMM:  C[M x Nreal] = A[M x K] * W[N x K]^T   (fp32 accumulate)
// Block tile 128x128, K-chunk 64, 8 waves (2x4), wave tile 64x32 (4x2 WMMA),
// 16 v_wmma per stage. Tiles staged into LDS by the Tensor Data Mover
// (wave 0 issues 2 DMA descriptors + s_wait_tensorcnt) when available.
// Requires: M % 128 == 0 (grid.y = M/128), K % 64 == 0, W row-padded to
// gridDim.x*128 rows. Stores guarded by col < Nreal.
// ---------------------------------------------------------------------------
__device__ __forceinline__ v16bf ldsFrag(const bf16r* rowp, int klo) {
    v8bf lo = *(const v8bf*)(rowp + klo);
    v8bf hi = *(const v8bf*)(rowp + klo + 16);
    v16bf f;
#pragma unroll
    for (int i = 0; i < 8; ++i) { f[i] = lo[i]; f[i + 8] = hi[i]; }
    return f;
}

__global__ __launch_bounds__(256) void k_gemm_bf16(
    const bf16r* __restrict__ A, const bf16r* __restrict__ W,
    float* __restrict__ C, int Nreal, int K, int ldc) {
    __shared__ __align__(16) bf16r As[128 * 64];
    __shared__ __align__(16) bf16r Bs[128 * 64];
    const int tid  = threadIdx.x;
    const int lane = tid & 31;
    const int wave = tid >> 5;
    const int wm   = wave >> 2;  // 0..1  (M direction, 64 rows each)
    const int wn   = wave & 3;   // 0..3  (N direction, 32 cols each)
    const int blockM = blockIdx.y * 128;
    const int blockN = blockIdx.x * 128;
    const int r   = lane & 15;
    const int klo = (lane & 16) ? 8 : 0;   // ISA 16-bit A/B fragment K-layout

    v8f acc[4][2] = {};

    for (int k0 = 0; k0 < K; k0 += 64) {
#if USE_TDM
        if (wave == 0) {
            tdm_load_2d_bf16(&A[(size_t)blockM * K + k0],
                             (unsigned)(size_t)&As[0], 64u, 128u, (unsigned)K);
            tdm_load_2d_bf16(&W[(size_t)blockN * K + k0],
                             (unsigned)(size_t)&Bs[0], 64u, 128u, (unsigned)K);
            __builtin_amdgcn_s_wait_tensorcnt(0);
        }
        __syncthreads();
#else
#pragma unroll
        for (int it = 0; it < 4; ++it) {
            int idx = tid + it * 256;     // 1024 chunks of 8 halves (16B) each
            int row = idx >> 3;
            int col = (idx & 7) << 3;
            *(v8bf*)&As[row * 64 + col] =
                *(const v8bf*)&A[(size_t)(blockM + row) * K + k0 + col];
            *(v8bf*)&Bs[row * 64 + col] =
                *(const v8bf*)&W[(size_t)(blockN + row) * K + k0 + col];
        }
        __syncthreads();
#endif
#pragma unroll
        for (int kk = 0; kk < 2; ++kk) {
            v16bf af[4], bfm[2];
#pragma unroll
            for (int tm = 0; tm < 4; ++tm)
                af[tm] = ldsFrag(&As[(wm * 64 + tm * 16 + r) * 64 + kk * 32], klo);
#pragma unroll
            for (int tn = 0; tn < 2; ++tn)
                bfm[tn] = ldsFrag(&Bs[(wn * 32 + tn * 16 + r) * 64 + kk * 32], klo);
#pragma unroll
            for (int tm = 0; tm < 4; ++tm)
#pragma unroll
                for (int tn = 0; tn < 2; ++tn)
                    acc[tm][tn] = __builtin_amdgcn_wmma_f32_16x16x32_bf16(
                        false, af[tm], false, bfm[tn], (short)0, acc[tm][tn],
                        false, false);
        }
        __syncthreads();
    }

    // C/D layout: VGPR j -> row = j + (lane>=16)*8, col = lane&15
    const int rbase = blockM + wm * 64 + ((lane >> 4) << 3);
    const int cbase = blockN + wn * 32 + (lane & 15);
#pragma unroll
    for (int tm = 0; tm < 4; ++tm) {
#pragma unroll
        for (int tn = 0; tn < 2; ++tn) {
            int cc = cbase + tn * 16;
            if (cc < Nreal) {
#pragma unroll
                for (int j = 0; j < 8; ++j)
                    C[(size_t)(rbase + tm * 16 + j) * ldc + cc] = acc[tm][tn][j];
            }
        }
    }
}

// ---------------------------------------------------------------------------
// Per (token, head) post-processing of in_proj output
// ---------------------------------------------------------------------------
__global__ __launch_bounds__(64) void k_post(
    const float* __restrict__ proj, const float* __restrict__ A_log,
    const float* __restrict__ Dv, const float* __restrict__ dt_bias,
    const float* __restrict__ B_bias, const float* __restrict__ C_bias,
    const float* __restrict__ Bnw, const float* __restrict__ Cnw,
    float* __restrict__ Bbuf, float* __restrict__ Cbuf,
    float* __restrict__ xsc, float* __restrict__ alpha,
    float* __restrict__ dtsp, float* __restrict__ skip) {
    const int h = blockIdx.x, tok = blockIdx.y, t = threadIdx.x;
    const float* p = proj + (size_t)tok * PROJ;
    float bv = p[OFF_B + h * DSTATE + t];
    float cv = p[OFF_C + h * DSTATE + t];
    float xv = p[OFF_X + h * HEADDIM + t];
    __shared__ float sB[64], sC[64], sX[64], sred[3];
    sB[t] = bv * bv; sC[t] = cv * cv; sX[t] = xv;
    __syncthreads();
    if (t == 0) {
        float a = 0.f, b = 0.f, c = 0.f;
        for (int i = 0; i < 64; ++i) { a += sB[i]; b += sC[i]; c += sX[i]; }
        sred[0] = a; sred[1] = b; sred[2] = c;
    }
    __syncthreads();
    float dtv = p[OFF_DT + h] + dt_bias[h];
    float sp  = (dtv > 20.f) ? dtv : log1pf(expf(dtv));
    float Aa  = -expf(A_log[h]) * sp;
    float al  = expf(Aa);
    float gm  = (al - 1.f) / (Aa + 1e-6f) * 0.5f + 1.f;
    size_t hb = (size_t)tok * NHEADS + h;
    float bsc = rsqrtf(sred[0] / (float)DSTATE + 1e-5f);
    float csc = rsqrtf(sred[1] / (float)DSTATE + 1e-5f);
    Bbuf[hb * DSTATE + t] = bv * bsc * Bnw[t] + B_bias[h * DSTATE + t];
    Cbuf[hb * DSTATE + t] = cv * csc * Cnw[t] + C_bias[h * DSTATE + t];
    xsc[hb * HEADDIM + t] = xv * gm;
    if (t == 0) { alpha[hb] = al; dtsp[hb] = sp; skip[hb] = Dv[h] * sred[2]; }
}

// ---------------------------------------------------------------------------
// Serial cumsum of dt*theta over S + in-place RoPE of B, C
// grid = B_*NHEADS, 32 threads (one rope pair each)
// ---------------------------------------------------------------------------
__global__ __launch_bounds__(32) void k_rope(
    const float* __restrict__ proj, const float* __restrict__ dtsp,
    float* __restrict__ Bbuf, float* __restrict__ Cbuf) {
    int b = blockIdx.x / NHEADS, h = blockIdx.x % NHEADS, d = threadIdx.x;
    float acc = 0.f;
    for (int s = 0; s < S_; ++s) {
        size_t tok = (size_t)b * S_ + s;
        size_t hb  = tok * NHEADS + h;
        acc += dtsp[hb] * proj[tok * PROJ + OFF_TH + h * (DSTATE / 2) + d];
        float cth = cosf(acc), sth = sinf(acc);
        size_t base = hb * DSTATE;
        float b1 = Bbuf[base + 2 * d], b2 = Bbuf[base + 2 * d + 1];
        Bbuf[base + 2 * d]     = cth * b1 - sth * b2;
        Bbuf[base + 2 * d + 1] = sth * b1 + cth * b2;
        float c1 = Cbuf[base + 2 * d], c2 = Cbuf[base + 2 * d + 1];
        Cbuf[base + 2 * d]     = cth * c1 - sth * c2;
        Cbuf[base + 2 * d + 1] = sth * c1 + cth * c2;
    }
}

// ---------------------------------------------------------------------------
// SSM scan: one workgroup per (b, h), 64x64 state split 16 elems/thread
// thread t: p = t>>2, n-chunk = (t&3)*16 ; quad shfl_xor reduce for y[p]
// ---------------------------------------------------------------------------
__global__ __launch_bounds__(256) void k_scan(
    const float* __restrict__ Bbuf, const float* __restrict__ Cbuf,
    const float* __restrict__ xsc, const float* __restrict__ alpha,
    const float* __restrict__ skip, float* __restrict__ y) {
    int b = blockIdx.x / NHEADS, h = blockIdx.x % NHEADS;
    int t = threadIdx.x, p = t >> 2, nc = t & 3;
    float st[16];
#pragma unroll
    for (int i = 0; i < 16; ++i) st[i] = 0.f;
    __shared__ float sB[64], sC[64], sX[64], sS[2];
    for (int s = 0; s < S_; ++s) {
        size_t hb = ((size_t)b * S_ + s) * NHEADS + h;
        if (t < 64)        sB[t]       = Bbuf[hb * DSTATE + t];
        else if (t < 128)  sC[t - 64]  = Cbuf[hb * DSTATE + (t - 64)];
        else if (t < 192)  sX[t - 128] = xsc[hb * HEADDIM + (t - 128)];
        else if (t == 192) { sS[0] = alpha[hb]; sS[1] = skip[hb]; }
        __syncthreads();
        float al = sS[0], xp = sX[p];
        float part = 0.f;
#pragma unroll
        for (int j = 0; j < 16; ++j) {
            int n = nc * 16 + j;
            st[j] = st[j] * al + sB[n] * xp;
            part += st[j] * sC[n];
        }
        part += __shfl_xor(part, 1, 32);
        part += __shfl_xor(part, 2, 32);
        if (nc == 0) y[hb * HEADDIM + p] = part + sS[1];
        __syncthreads();
    }
}

// ---------------------------------------------------------------------------
// Elementwise kernels
// ---------------------------------------------------------------------------
__global__ void k_gate(const float* __restrict__ proj,
                       const float* __restrict__ y, bf16r* __restrict__ yg) {
    size_t i = (size_t)blockIdx.x * 256 + threadIdx.x;
    if (i >= (size_t)TOK * DINNER) return;
    size_t tok = i / DINNER;
    int c = (int)(i % DINNER);
    float z = proj[tok * PROJ + OFF_Z + c];
    yg[i] = f2bf(y[i] * z / (1.f + expf(-z)));
}

__global__ void k_add2(const float* __restrict__ a, const float* __restrict__ b,
                       float* __restrict__ o, size_t n) {
    size_t i = (size_t)blockIdx.x * 256 + threadIdx.x;
    if (i < n) o[i] = a[i] + b[i];
}

__global__ void k_silu_mul(const float* __restrict__ g,
                           const float* __restrict__ u,
                           bf16r* __restrict__ m, size_t n) {
    size_t i = (size_t)blockIdx.x * 256 + threadIdx.x;
    if (i >= n) return;
    float gv = g[i];
    m[i] = f2bf(gv / (1.f + expf(-gv)) * u[i]);
}

// ---------------------------------------------------------------------------
extern "C" void kernel_launch(void* const* d_in, const int* in_sizes, int n_in,
                              void* d_out, int out_size, void* d_ws, size_t ws_size,
                              hipStream_t stream) {
    const float* x          = (const float*)d_in[0];
    const float* norm1_w    = (const float*)d_in[1];
    const float* norm2_w    = (const float*)d_in[2];
    const float* in_proj_w  = (const float*)d_in[3];
    const float* out_proj_w = (const float*)d_in[4];
    const float* A_log      = (const float*)d_in[5];
    const float* Dv         = (const float*)d_in[6];
    const float* dt_bias    = (const float*)d_in[7];
    const float* B_bias     = (const float*)d_in[8];
    const float* C_bias     = (const float*)d_in[9];
    const float* Bnw        = (const float*)d_in[10];
    const float* Cnw        = (const float*)d_in[11];
    const float* mlp_gate_w = (const float*)d_in[12];
    const float* mlp_up_w   = (const float*)d_in[13];
    const float* mlp_down_w = (const float*)d_in[14];
    float* out = (float*)d_out;

    char* ws = (char*)d_ws;
    size_t off = 0;
    auto alloc = [&](size_t bytes) -> void* {
        off = (off + 255) & ~(size_t)255;
        void* p = ws + off;
        off += bytes;
        return p;
    };

    bf16r* w_in   = (bf16r*)alloc((size_t)PROJ_PAD * D_ * 2);
    bf16r* w_out  = (bf16r*)alloc((size_t)D_ * DINNER * 2);
    bf16r* w_gate = (bf16r*)alloc((size_t)DMLP * D_ * 2);
    bf16r* w_up   = (bf16r*)alloc((size_t)DMLP * D_ * 2);
    bf16r* w_down = (bf16r*)alloc((size_t)D_ * DMLP * 2);
    bf16r* h1     = (bf16r*)alloc((size_t)TOK * D_ * 2);
    float* proj   = (float*)alloc((size_t)TOK * PROJ * 4);
    float* Bbuf   = (float*)alloc((size_t)TOK * NHEADS * DSTATE * 4);
    float* Cbuf   = (float*)alloc((size_t)TOK * NHEADS * DSTATE * 4);
    float* xsc    = (float*)alloc((size_t)TOK * NHEADS * HEADDIM * 4);
    float* alpha  = (float*)alloc((size_t)TOK * NHEADS * 4);
    float* dtsp   = (float*)alloc((size_t)TOK * NHEADS * 4);
    float* skip   = (float*)alloc((size_t)TOK * NHEADS * 4);
    float* ybuf   = (float*)alloc((size_t)TOK * DINNER * 4);
    bf16r* yg     = (bf16r*)alloc((size_t)TOK * DINNER * 2);
    float* delta  = (float*)alloc((size_t)TOK * D_ * 4);
    float* x2     = (float*)alloc((size_t)TOK * D_ * 4);
    bf16r* h2     = (bf16r*)alloc((size_t)TOK * D_ * 2);
    float* gbuf   = (float*)alloc((size_t)TOK * DMLP * 4);
    float* ubuf   = (float*)alloc((size_t)TOK * DMLP * 4);
    bf16r* mbuf   = (bf16r*)alloc((size_t)TOK * DMLP * 2);
    float* mlpd   = (float*)alloc((size_t)TOK * D_ * 4);

    auto cdiv = [](size_t a, size_t b) { return (unsigned)((a + b - 1) / b); };

    // 1. Weight conversion (fp32 -> bf16, row-padded)
    k_cvt_pad<<<cdiv((size_t)PROJ_PAD * D_, 256), 256, 0, stream>>>(in_proj_w,  w_in,   PROJ, D_,   PROJ_PAD);
    k_cvt_pad<<<cdiv((size_t)D_ * DINNER,  256), 256, 0, stream>>>(out_proj_w, w_out,  D_,   DINNER, D_);
    k_cvt_pad<<<cdiv((size_t)DMLP * D_,    256), 256, 0, stream>>>(mlp_gate_w, w_gate, DMLP, D_,   DMLP);
    k_cvt_pad<<<cdiv((size_t)DMLP * D_,    256), 256, 0, stream>>>(mlp_up_w,   w_up,   DMLP, D_,   DMLP);
    k_cvt_pad<<<cdiv((size_t)D_ * DMLP,    256), 256, 0, stream>>>(mlp_down_w, w_down, D_,   DMLP, D_);

    // 2. norm1 -> h1 (bf16), in_proj GEMM -> proj
    k_rmsnorm_bf16<<<TOK, 256, 0, stream>>>(x, norm1_w, h1, D_);
    k_gemm_bf16<<<dim3(PROJ_PAD / 128, TOK / 128), 256, 0, stream>>>(h1, w_in, proj, PROJ, D_, PROJ);

    // 3. per-head post-processing, cumsum+rope, scan
    k_post<<<dim3(NHEADS, TOK), 64, 0, stream>>>(proj, A_log, Dv, dt_bias, B_bias,
                                                 C_bias, Bnw, Cnw, Bbuf, Cbuf,
                                                 xsc, alpha, dtsp, skip);
    k_rope<<<B_ * NHEADS, 32, 0, stream>>>(proj, dtsp, Bbuf, Cbuf);
    k_scan<<<B_ * NHEADS, 256, 0, stream>>>(Bbuf, Cbuf, xsc, alpha, skip, ybuf);

    // 4. gate with silu(z), out_proj GEMM, residual
    k_gate<<<cdiv((size_t)TOK * DINNER, 256), 256, 0, stream>>>(proj, ybuf, yg);
    k_gemm_bf16<<<dim3(D_ / 128, TOK / 128), 256, 0, stream>>>(yg, w_out, delta, D_, DINNER, D_);
    k_add2<<<cdiv((size_t)TOK * D_, 256), 256, 0, stream>>>(x, delta, x2, (size_t)TOK * D_);

    // 5. MLP
    k_rmsnorm_bf16<<<TOK, 256, 0, stream>>>(x2, norm2_w, h2, D_);
    k_gemm_bf16<<<dim3(DMLP / 128, TOK / 128), 256, 0, stream>>>(h2, w_gate, gbuf, DMLP, D_, DMLP);
    k_gemm_bf16<<<dim3(DMLP / 128, TOK / 128), 256, 0, stream>>>(h2, w_up,   ubuf, DMLP, D_, DMLP);
    k_silu_mul<<<cdiv((size_t)TOK * DMLP, 256), 256, 0, stream>>>(gbuf, ubuf, mbuf, (size_t)TOK * DMLP);
    k_gemm_bf16<<<dim3(D_ / 128, TOK / 128), 256, 0, stream>>>(mbuf, w_down, mlpd, D_, DMLP, D_);
    k_add2<<<cdiv((size_t)TOK * D_, 256), 256, 0, stream>>>(x2, mlpd, out, (size_t)TOK * D_);
}
